// LayerRecurrent_46737834115725
// MI455X (gfx1250) — matrix operand order repaired and checked by reading, not compiled, not run on hardware
//
#include <hip/hip_runtime.h>
#include <hip/hip_bf16.h>

// ---------------------------------------------------------------------------
// ESN reservoir scan, fused, persistent-kernel, bf16 WMMA w/ fp32 accumulate.
//   h_t = 0.5*h_{t-1} + 0.5*tanh(batch_t @ W_in^T + h_{t-1} @ W_res^T + bias)
// out[B,T,D] fp32.
//
// - W_res / W_in slices live in LDS (bf16) for all 512 steps.
// - fp32 h state lives in registers (static tile ownership); only the bf16
//   copy (WMMA A-operand for all waves) ping-pongs through global/L2.
// - Input projection of step t+1 is computed between barrier signal and
//   barrier wait, hiding it behind grid-barrier skew.
// ---------------------------------------------------------------------------

typedef __attribute__((ext_vector_type(16))) __bf16 v16bf;
typedef __attribute__((ext_vector_type(8)))  __bf16 v8bf;
typedef __attribute__((ext_vector_type(8)))  float  v8f;

#define T_STEPS 512
#define BATCH   64
#define IDIM    768
#define DDIM    2048
#define LEAK    0.5f

#define NWG      64            // one WG per 32 output columns
#define NTHREADS 256           // 8 wave32
#define N_PER_WG 32
#define WSTRIDE  (DDIM + 8)    // padded LDS row stride (bf16) -> conflict-free b128
#define USTRIDE  (IDIM + 8)

#define HSIZE      (BATCH * DDIM)
#define WS_HBF_OFF 1024        // __bf16 hbf[2][BATCH][DDIM]

__global__ void esn_init(__bf16* __restrict__ hbf, unsigned* __restrict__ cnt) {
    int i = blockIdx.x * blockDim.x + threadIdx.x;
    if (i == 0) *cnt = 0u;
    if (i < HSIZE) hbf[i] = (__bf16)0.0f;
}

__device__ __forceinline__ v16bf cat16(v8bf a, v8bf b) {
    return __builtin_shufflevector(a, b, 0,1,2,3,4,5,6,7,8,9,10,11,12,13,14,15);
}

// acc = bias + batch_t(16x768 tile) @ W_in^T slice  (independent of h)
__device__ __forceinline__ v8f input_proj(const float* __restrict__ batch,
                                          const __bf16* __restrict__ Urow,
                                          int t, int mrow, int hi, float bv) {
    v8f acc = {bv, bv, bv, bv, bv, bv, bv, bv};
    const float* arow = batch + ((size_t)t * BATCH + mrow) * IDIM;
    #pragma unroll 2
    for (int kk = 0; kk < IDIM; kk += 32) {
        const float* p0 = arow + kk + hi * 8;         // A elems 0..7
        const float* p1 = p0 + 16;                    // A elems 8..15
        float4 f0 = *(const float4*)(p0);
        float4 f1 = *(const float4*)(p0 + 4);
        float4 f2 = *(const float4*)(p1);
        float4 f3 = *(const float4*)(p1 + 4);
        v16bf A;
        A[0]=(__bf16)f0.x;  A[1]=(__bf16)f0.y;  A[2]=(__bf16)f0.z;  A[3]=(__bf16)f0.w;
        A[4]=(__bf16)f1.x;  A[5]=(__bf16)f1.y;  A[6]=(__bf16)f1.z;  A[7]=(__bf16)f1.w;
        A[8]=(__bf16)f2.x;  A[9]=(__bf16)f2.y;  A[10]=(__bf16)f2.z; A[11]=(__bf16)f2.w;
        A[12]=(__bf16)f3.x; A[13]=(__bf16)f3.y; A[14]=(__bf16)f3.z; A[15]=(__bf16)f3.w;
        const __bf16* bp = Urow + kk + hi * 8;        // B elem j -> k = kk+8*hi+j (j<8), kk+16+8*hi+(j-8)
        v16bf Bf = cat16(*(const v8bf*)bp, *(const v8bf*)(bp + 16));
        acc = __builtin_amdgcn_wmma_f32_16x16x32_bf16(false, A, false, Bf,
                                                      (short)0, acc, false, false);
    }
    return acc;
}

__global__ void __launch_bounds__(NTHREADS, 1)
esn_scan(const float* __restrict__ batch, const float* __restrict__ inw,
         const float* __restrict__ resw,  const float* __restrict__ bias,
         float* __restrict__ out, __bf16* __restrict__ hbf,
         unsigned* __restrict__ cnt)
{
    extern __shared__ char smem[];
    __bf16* Wsl = (__bf16*)smem;                                       // [32][WSTRIDE]
    __bf16* Usl = (__bf16*)(smem + (size_t)N_PER_WG * WSTRIDE * 2);    // [32][USTRIDE]

    const int wg   = blockIdx.x;
    const int tid  = threadIdx.x;
    const int wave = tid >> 5;
    const int lane = tid & 31;
    const int mt   = wave & 3;          // 4 row tiles of 16 (batch dim 64)
    const int ntl  = wave >> 2;         // 2 col tiles of 16 per WG
    const int hi   = lane >> 4;         // half-wave select (ISA A/B K-split)
    const int lm   = lane & 15;
    const int nbaseWG = wg * N_PER_WG;
    const int nloc = ntl * 16 + lm;     // local output column 0..31
    const int n    = nbaseWG + nloc;    // global output column
    const int mrow = mt * 16 + lm;      // A-matrix row (batch index)

    // ---- Stage this WG's weight slices into LDS as bf16 (once; reused 512x)
    for (int idx = tid; idx < N_PER_WG * DDIM; idx += NTHREADS) {
        int r = idx >> 11, k = idx & (DDIM - 1);
        Wsl[r * WSTRIDE + k] = (__bf16)resw[(size_t)(nbaseWG + r) * DDIM + k];
    }
    for (int idx = tid; idx < N_PER_WG * IDIM; idx += NTHREADS) {
        int r = idx / IDIM, k = idx - r * IDIM;
        Usl[r * USTRIDE + k] = (__bf16)inw[(size_t)(nbaseWG + r) * IDIM + k];
    }
    __syncthreads();

    const float   bv   = bias[n];
    const __bf16* Wrow = Wsl + (size_t)nloc * WSTRIDE;
    const __bf16* Urow = Usl + (size_t)nloc * USTRIDE;

    // fp32 recurrent state for this wave's tile lives in registers:
    // lane -> column n; element r -> row m = mt*16 + 8*hi + r.
    v8f hprev = {0.f, 0.f, 0.f, 0.f, 0.f, 0.f, 0.f, 0.f};

    // Input projection for step 0 (h-independent).
    v8f accIn = input_proj(batch, Urow, 0, mrow, hi, bv);

    for (int t = 0; t < T_STEPS; ++t) {
        const int rd = t & 1, wr = rd ^ 1;           // ping-pong bf16 h buffers
        v8f acc = accIn;

        // ---- recurrent: acc += h[16x2048] @ W_res^T slice (LDS-resident B)
        const __bf16* hrow = hbf + (size_t)rd * HSIZE + (size_t)mrow * DDIM;
        #pragma unroll 8
        for (int kk = 0; kk < DDIM; kk += 32) {
            v8bf a0 = *(const v8bf*)(hrow + kk + hi * 8);
            v8bf a1 = *(const v8bf*)(hrow + kk + 16 + hi * 8);
            v16bf A = cat16(a0, a1);
            const __bf16* bp = Wrow + kk + hi * 8;
            v16bf Bf = cat16(*(const v8bf*)bp, *(const v8bf*)(bp + 16));
            acc = __builtin_amdgcn_wmma_f32_16x16x32_bf16(false, A, false, Bf,
                                                          (short)0, acc, false, false);
        }

        // ---- leaky-tanh update; C layout: lane -> n = lane&15, m = r + 8*(lane>>4)
        __bf16* hbw = hbf + (size_t)wr * HSIZE;
        #pragma unroll
        for (int r = 0; r < 8; ++r) {
            int m = mt * 16 + hi * 8 + r;
            float hn = (1.0f - LEAK) * hprev[r] + LEAK * tanhf(acc[r]);
            hprev[r] = hn;
            hbw[(size_t)m * DDIM + n] = (__bf16)hn;
            out[((size_t)m * T_STEPS + t) * DDIM + n] = hn;
        }

        if (t + 1 < T_STEPS) {
            // ---- split device-wide barrier: signal, overlap, wait
            __threadfence();                         // release this thread's stores
            __syncthreads();
            if (tid == 0)
                __hip_atomic_fetch_add(cnt, 1u, __ATOMIC_RELEASE,
                                       __HIP_MEMORY_SCOPE_AGENT);

            // Overlapped work (h-independent): next step's input projection.
            accIn = input_proj(batch, Urow, t + 1, mrow, hi, bv);

            if (tid == 0) {
                const unsigned target = (unsigned)(t + 1) * NWG;
                while (__hip_atomic_load(cnt, __ATOMIC_ACQUIRE,
                                         __HIP_MEMORY_SCOPE_AGENT) < target)
                    __builtin_amdgcn_s_sleep(1);
            }
            __syncthreads();
        }
    }
}

extern "C" void kernel_launch(void* const* d_in, const int* in_sizes, int n_in,
                              void* d_out, int out_size, void* d_ws, size_t ws_size,
                              hipStream_t stream) {
    const float* batch = (const float*)d_in[0];  // [512,64,768]
    const float* inw   = (const float*)d_in[1];  // [2048,768]
    const float* resw  = (const float*)d_in[2];  // [2048,2048]
    const float* bias  = (const float*)d_in[3];  // [2048]
    float* out = (float*)d_out;                  // [64,512,2048]

    unsigned* cnt = (unsigned*)d_ws;
    __bf16*   hbf = (__bf16*)((char*)d_ws + WS_HBF_OFF);

    esn_init<<<(HSIZE + 255) / 256, 256, 0, stream>>>(hbf, cnt);

    const size_t smemBytes = (size_t)(N_PER_WG * WSTRIDE + N_PER_WG * USTRIDE) * sizeof(__bf16);
    hipFuncSetAttribute(reinterpret_cast<const void*>(esn_scan),
                        hipFuncAttributeMaxDynamicSharedMemorySize, (int)smemBytes);
    esn_scan<<<NWG, NTHREADS, smemBytes, stream>>>(batch, inw, resw, bias,
                                                   out, hbf, cnt);
}
